// MyClassifier_38817914421544
// MI455X (gfx1250) — matrix operand order repaired and checked by reading, not compiled, not run on hardware
//
#include <hip/hip_runtime.h>
#include <hip/hip_bf16.h>
#include <math.h>

typedef __attribute__((ext_vector_type(16))) _Float16 v16h;
typedef __attribute__((ext_vector_type(8)))  _Float16 v8h;
typedef __attribute__((ext_vector_type(8)))  float    v8f;
typedef __attribute__((ext_vector_type(4)))  int      v4i;

#define Bsz 16
#define Ssz 2048
#define Tsz 24
#define Csz 16
#define Lsz 8
#define Dsz 1024
#define Hsz 512
#define G4  2048

// ---- CDNA5 async global->LDS path (probe-guarded) --------------------------
#if defined(__has_builtin)
#if __has_builtin(__builtin_amdgcn_global_load_async_to_lds_b128) && \
    __has_builtin(__builtin_amdgcn_s_wait_asynccnt)
#define USE_ASYNC_LDS 1
#endif
#if __has_builtin(__builtin_amdgcn_sched_group_barrier)
#define HAVE_SCHED_GROUPS 1
#endif
#endif
#ifndef USE_ASYNC_LDS
#define USE_ASYNC_LDS 0
#endif
#ifndef HAVE_SCHED_GROUPS
#define HAVE_SCHED_GROUPS 0
#endif

#if USE_ASYNC_LDS
typedef __attribute__((address_space(1))) v4i* gas_t;
typedef __attribute__((address_space(3))) v4i* las_t;
// copy 32 bytes (one thread) global -> LDS, async (2 x b128)
#define TILE_CP(g, l)                                                          \
    do {                                                                       \
        __builtin_amdgcn_global_load_async_to_lds_b128((gas_t)(g), (las_t)(l), \
                                                       0, 0);                  \
        __builtin_amdgcn_global_load_async_to_lds_b128((gas_t)(g), (las_t)(l), \
                                                       16, 0);                 \
    } while (0)
#define TILE_WAIT(n) __builtin_amdgcn_s_wait_asynccnt(n)
#else
#define TILE_CP(g, l)                                                          \
    do {                                                                       \
        v8h t0 = *(const v8h*)(g);                                             \
        v8h t1 = *(const v8h*)((const _Float16*)(g) + 8);                      \
        *(v8h*)(l) = t0;                                                       \
        *(v8h*)((_Float16*)(l) + 8) = t1;                                      \
    } while (0)
#define TILE_WAIT(n)
#endif

#if HAVE_SCHED_GROUPS
// Per K-chunk issue order: 2 global A loads -> 8 LDS B reads -> 4 WMMA.
// Forces RA to keep 4 distinct B-fragment register groups alive, so the four
// v_wmma issue back-to-back behind a single dscnt wait.
#define SCHED_PIN()                                                   \
    do {                                                              \
        __builtin_amdgcn_sched_group_barrier(0x020, 2, 0); /* VMEM rd */ \
        __builtin_amdgcn_sched_group_barrier(0x100, 8, 0); /* DS rd  */ \
        __builtin_amdgcn_sched_group_barrier(0x008, 4, 0); /* WMMA   */ \
    } while (0)
#else
#define SCHED_PIN()
#endif

#define KSTEP 64
#define LDS_STRIDE 72  // 64 f16 + 8 pad, keeps 16B alignment, spreads banks

// ---------------------------------------------------------------------------
// WMMA GEMM: C[M,N] = act( A[M,K](f16,lda) @ W[N,K]^T + bias )
// Requirements: N % 64 == 0, K % 64 == 0, M % 16 == 0 (launches use M%128==0).
// Block = 256 thr = 8 waves; block computes 128(M) x 64(N); B tile staged in
// LDS (double-buffered, async on CDNA5); each wave does a 16x64 strip.
// ---------------------------------------------------------------------------
__global__ __launch_bounds__(256)
void gemm_wmma_kernel(const _Float16* __restrict__ A, int lda,
                      const _Float16* __restrict__ W, int ldw,
                      const float* __restrict__ bias,
                      float* __restrict__ C, int ldc, int accum,
                      _Float16* __restrict__ Ch, int ldch,
                      int M, int N, int K, int act)
{
    __shared__ _Float16 bt[2][64 * LDS_STRIDE];
    int tid  = threadIdx.x;
    int wave = tid >> 5;
    int lane = tid & 31;
    int mt   = blockIdx.y * 8 + wave;
    int n0   = blockIdx.x * 64;
    bool mvalid = (mt * 16 < M);

    // A fragment (16-bit A 16x32): lane m = lane&15,
    // lanes 0-15 -> K 0..15 ; lanes 16-31 -> K 16..31
    int am  = lane & 15;
    int akb = (lane >> 4) << 4;
    // B fragment: lane n = lane&15, 16 contiguous K per lane half.
    int bn  = lane & 15;
    int bkb = (lane >> 4) << 4;
    const _Float16* arow = A + (size_t)(mt * 16 + am) * lda + akb;

    // cooperative B-tile loader: thread -> (row 0..63, 32B chunk 0..3)
    int lr = tid >> 2;
    int lc = (tid & 3) * 16;
    const _Float16* gsrc = W + (size_t)(n0 + lr) * ldw + lc;
    _Float16* lb0 = &bt[0][lr * LDS_STRIDE + lc];
    _Float16* lb1 = &bt[1][lr * LDS_STRIDE + lc];

    int nstage = K / KSTEP;
    TILE_CP(gsrc, lb0);  // preload stage 0

    v8f acc[4] = {};
    for (int s = 0; s < nstage; ++s) {
        _Float16* lnext = (s & 1) ? lb0 : lb1;
        const _Float16* lcur = (s & 1) ? &bt[1][0] : &bt[0][0];
        if (s + 1 < nstage) {
            TILE_CP(gsrc + (size_t)(s + 1) * KSTEP, lnext);
            TILE_WAIT(2);  // stage-s data (issued last iter) has landed
        } else {
            TILE_WAIT(0);
        }
        __syncthreads();
        if (mvalid) {
            const _Float16* ap = arow + (size_t)s * KSTEP;
            if (s + 1 < nstage)
                __builtin_prefetch(ap + KSTEP, 0, 3);  // global_prefetch_b8
#pragma unroll
            for (int kc = 0; kc < 2; ++kc) {
                union { v16h v; v8h h[2]; } ua;
                ua.h[0] = *(const v8h*)(ap + kc * 32);
                ua.h[1] = *(const v8h*)(ap + kc * 32 + 8);
                // load all 4 B fragments first (distinct regs -> one dscnt
                // wait, WMMAs issue back-to-back)
                const _Float16* bp = lcur + bn * LDS_STRIDE + bkb + kc * 32;
                union { v16h v; v8h h[2]; } ub0, ub1, ub2, ub3;
                ub0.h[0] = *(const v8h*)(bp);
                ub0.h[1] = *(const v8h*)(bp + 8);
                ub1.h[0] = *(const v8h*)(bp + 16 * LDS_STRIDE);
                ub1.h[1] = *(const v8h*)(bp + 16 * LDS_STRIDE + 8);
                ub2.h[0] = *(const v8h*)(bp + 32 * LDS_STRIDE);
                ub2.h[1] = *(const v8h*)(bp + 32 * LDS_STRIDE + 8);
                ub3.h[0] = *(const v8h*)(bp + 48 * LDS_STRIDE);
                ub3.h[1] = *(const v8h*)(bp + 48 * LDS_STRIDE + 8);
                acc[0] = __builtin_amdgcn_wmma_f32_16x16x32_f16(
                    false, ua.v, false, ub0.v, (short)0, acc[0], false, false);
                acc[1] = __builtin_amdgcn_wmma_f32_16x16x32_f16(
                    false, ua.v, false, ub1.v, (short)0, acc[1], false, false);
                acc[2] = __builtin_amdgcn_wmma_f32_16x16x32_f16(
                    false, ua.v, false, ub2.v, (short)0, acc[2], false, false);
                acc[3] = __builtin_amdgcn_wmma_f32_16x16x32_f16(
                    false, ua.v, false, ub3.v, (short)0, acc[3], false, false);
                SCHED_PIN();
            }
        }
        __syncthreads();  // done reading lcur; next iter may overwrite it
    }

    if (!mvalid) return;
    int laneh = lane >> 4;
    int nl    = lane & 15;
#pragma unroll
    for (int j = 0; j < 4; ++j) {
        int n = n0 + j * 16 + nl;
        float bv = bias ? bias[n] : 0.0f;
#pragma unroll
        for (int i = 0; i < 8; ++i) {
            int m = mt * 16 + laneh * 8 + i;  // D-matrix VGPR layout
            float v = acc[j][i] + bv;
            if (accum) v += C[(size_t)m * ldc + n];
            if (act == 1) v = (v >= 0.0f) ? v : 0.01f * v;
            if (C)  C [(size_t)m * ldc  + n] = v;
            if (Ch) Ch[(size_t)m * ldch + n] = (_Float16)v;
        }
    }
}

// ---------------------------------------------------------------------------
// Small-N WMMA GEMM (N == 16): logits head. Direct global loads, one tile.
// ---------------------------------------------------------------------------
__global__ __launch_bounds__(256)
void gemm_small_kernel(const _Float16* __restrict__ A, int lda,
                       const _Float16* __restrict__ W, int ldw,
                       const float* __restrict__ bias,
                       float* __restrict__ C, int ldc, int M, int K)
{
    int wave = threadIdx.x >> 5;
    int lane = threadIdx.x & 31;
    int mt   = blockIdx.y * 8 + wave;
    if (mt * 16 >= M) return;
    int am  = lane & 15;
    int akb = (lane >> 4) << 4;
    int bn  = lane & 15;
    int bkb = (lane >> 4) << 4;
    const _Float16* arow = A + (size_t)(mt * 16 + am) * lda + akb;
    v8f acc = {};
    for (int k0 = 0; k0 < K; k0 += 32) {
        union { v16h v; v8h h[2]; } ua;
        ua.h[0] = *(const v8h*)(arow + k0);
        ua.h[1] = *(const v8h*)(arow + k0 + 8);
        const v16h b = *(const v16h*)(W + (size_t)bn * ldw + bkb + k0);
        acc = __builtin_amdgcn_wmma_f32_16x16x32_f16(false, ua.v, false, b,
                                                     (short)0, acc, false,
                                                     false);
    }
    int laneh = lane >> 4;
    int nl    = lane & 15;
    float bv  = bias ? bias[nl] : 0.0f;
#pragma unroll
    for (int i = 0; i < 8; ++i) {
        int m = mt * 16 + laneh * 8 + i;
        C[(size_t)m * ldc + nl] = acc[i] + bv;
    }
}

// ---------------------------------------------------------------------------
__global__ void gather_kernel(const float* __restrict__ seq,
                              const int* __restrict__ ids,
                              _Float16* __restrict__ X,
                              int spans_per_b)
{
    int r  = blockIdx.x;  // r = n*L + l
    int n  = r >> 3;      // L == 8
    int b  = n / spans_per_b;
    int id = ids[r];
    const float* src = seq + ((size_t)b * Ssz + id) * Dsz;
    _Float16*    dst = X + (size_t)r * Dsz;
    for (int j = threadIdx.x; j < Dsz; j += blockDim.x)
        dst[j] = (_Float16)src[j];
}

__global__ void lstm_cell_kernel(const float* __restrict__ gates,
                                 float* __restrict__ cst,
                                 _Float16* __restrict__ hout, int ld_h,
                                 _Float16* __restrict__ hseq, int ld_seq,
                                 int Nn, int first)
{
    int idx = blockIdx.x * blockDim.x + threadIdx.x;
    if (idx >= Nn * Hsz) return;
    int n = idx >> 9, j = idx & 511;
    const float* g = gates + (size_t)n * G4;
    float gi = g[j], gf = g[Hsz + j], gg = g[2 * Hsz + j], go = g[3 * Hsz + j];
    float cp = first ? 0.0f : cst[idx];
    float si = 1.0f / (1.0f + __expf(-gi));
    float sf = 1.0f / (1.0f + __expf(-gf));
    float so = 1.0f / (1.0f + __expf(-go));
    float cn = sf * cp + si * tanhf(gg);
    float hn = so * tanhf(cn);
    cst[idx] = cn;
    hout[(size_t)n * ld_h + j] = (_Float16)hn;
    if (hseq) hseq[(size_t)n * ld_seq + j] = (_Float16)hn;
}

__global__ __launch_bounds__(256)
void pool_l2n_kernel(const float* __restrict__ name,
                     const float* __restrict__ comm,
                     const float* __restrict__ eng,
                     float* __restrict__ outf,
                     _Float16* __restrict__ outh,
                     int mode)
{
    int n = blockIdx.x;
    __shared__ float red[256];
    float vals[4];
    float ss = 0.0f;
#pragma unroll
    for (int i = 0; i < 4; ++i) {
        int j = threadIdx.x + i * 256;
        size_t o = (size_t)n * Dsz + j;
        float nm = name[o], cm = comm[o];
        float v;
        if (mode == 0) {
            float w = 1.0f / (1.0f + __expf(-eng[o]));
            v = nm + w * nm + (1.0f - w) * cm;
        } else {
            v = nm + cm;
        }
        vals[i] = v;
        ss += v * v;
    }
    red[threadIdx.x] = ss;
    __syncthreads();
    for (int s = 128; s > 0; s >>= 1) {
        if (threadIdx.x < s) red[threadIdx.x] += red[threadIdx.x + s];
        __syncthreads();
    }
    float inv = 1.0f / fmaxf(sqrtf(red[0]), 1e-12f);
#pragma unroll
    for (int i = 0; i < 4; ++i) {
        int j = threadIdx.x + i * 256;
        size_t o = (size_t)n * Dsz + j;
        float v = vals[i] * inv;
        outf[o] = v;
        outh[o] = (_Float16)v;
    }
}

__global__ __launch_bounds__(128)
void attn_kernel(const float* __restrict__ q,
                 const float* __restrict__ kk,
                 const float* __restrict__ vv,
                 _Float16* __restrict__ ctx)
{
    int nt = blockIdx.x >> 3;
    int h  = blockIdx.x & 7;
    int d  = threadIdx.x;
    __shared__ float red[128];
    __shared__ float sc[16];
    float qv = q[(size_t)nt * Dsz + h * 128 + d];
#pragma unroll 1
    for (int c = 0; c < Csz; ++c) {
        float p = qv * kk[((size_t)nt * Csz + c) * Dsz + h * 128 + d];
        red[d] = p;
        __syncthreads();
        for (int s = 64; s > 0; s >>= 1) {
            if (d < s) red[d] += red[d + s];
            __syncthreads();
        }
        if (d == 0) sc[c] = red[0] * 0.08838834764831845f;  // 128^-0.5
        __syncthreads();
    }
    float mx = -1e30f;
    for (int c = 0; c < Csz; ++c) mx = fmaxf(mx, sc[c]);
    float a[16], sum = 0.0f;
    for (int c = 0; c < Csz; ++c) { a[c] = __expf(sc[c] - mx); sum += a[c]; }
    float is = 1.0f / sum;
    float cd = 0.0f;
    for (int c = 0; c < Csz; ++c)
        cd += a[c] * is * vv[((size_t)nt * Csz + c) * Dsz + h * 128 + d];
    ctx[(size_t)nt * Dsz + h * 128 + d] = (_Float16)cd;
}

__global__ void cvt_f16_kernel(const float* __restrict__ s,
                               _Float16* __restrict__ d, long long n)
{
    long long i = (long long)blockIdx.x * blockDim.x + threadIdx.x;
    if (i < n) d[i] = (_Float16)s[i];
}

__global__ void pad_cls2_kernel(const float* __restrict__ W,
                                const float* __restrict__ b,
                                _Float16* __restrict__ Wp,
                                float* __restrict__ bp)
{
    int i = blockIdx.x * blockDim.x + threadIdx.x;
    if (i < 16 * 256) {
        int r = i >> 8;
        Wp[i] = (r < 2) ? (_Float16)W[i] : (_Float16)0.0f;
    }
    if (i < 16) bp[i] = (i < 2) ? b[i] : 0.0f;
}

__global__ void pack_kernel(const float* __restrict__ tl,
                            const float* __restrict__ cl,
                            float* __restrict__ out)
{
    int i = blockIdx.x * blockDim.x + threadIdx.x;
    if (i >= Bsz * Tsz * 17 * 2) return;
    int j    = i & 1;
    int slot = (i >> 1) % 17;
    int bt   = (i >> 1) / 17;
    out[i] = (slot == 0) ? tl[bt * 16 + j]
                         : cl[(bt * Csz + slot - 1) * 16 + j];
}

// ---------------------------------------------------------------------------
extern "C" void kernel_launch(void* const* d_in, const int* in_sizes, int n_in,
                              void* d_out, int out_size, void* d_ws, size_t ws_size,
                              hipStream_t stream)
{
    const float* seq        = (const float*)d_in[0];
    const float* lstm_Wih   = (const float*)d_in[1];
    const float* lstm_Whh   = (const float*)d_in[2];
    const float* lstm_b     = (const float*)d_in[3];
    const float* after_W    = (const float*)d_in[4];
    const float* after_b    = (const float*)d_in[5];
    const float* gate_W     = (const float*)d_in[6];
    const float* gate_b     = (const float*)d_in[7];
    const float* energy_W   = (const float*)d_in[8];
    const float* energy_b   = (const float*)d_in[9];
    const float* attn_in_W  = (const float*)d_in[10];
    const float* attn_in_b  = (const float*)d_in[11];
    const float* attn_out_W = (const float*)d_in[12];
    const float* attn_out_b = (const float*)d_in[13];
    const float* cls_t_W1   = (const float*)d_in[14];
    const float* cls_t_b1   = (const float*)d_in[15];
    const float* cls_t_W2   = (const float*)d_in[16];
    const float* cls_t_b2   = (const float*)d_in[17];
    const float* cls_c_W1   = (const float*)d_in[18];
    const float* cls_c_b1   = (const float*)d_in[19];
    const float* cls_c_W2   = (const float*)d_in[20];
    const float* cls_c_b2   = (const float*)d_in[21];

    char*  base = (char*)d_ws;
    size_t off  = 0;
    auto alloc = [&](size_t bytes) -> void* {
        void* p = base + off;
        off = (off + bytes + 255) & ~(size_t)255;
        return p;
    };

    _Float16* wih16    = (_Float16*)alloc((size_t)16777216 * 2);
    _Float16* whh16    = (_Float16*)alloc((size_t)8388608 * 2);
    _Float16* after16  = (_Float16*)alloc((size_t)4194304 * 2);
    _Float16* gate16   = (_Float16*)alloc((size_t)4194304 * 2);
    _Float16* energy16 = (_Float16*)alloc((size_t)4194304 * 2);
    _Float16* ain16    = (_Float16*)alloc((size_t)3145728 * 2);
    _Float16* aout16   = (_Float16*)alloc((size_t)1048576 * 2);
    _Float16* ct116    = (_Float16*)alloc((size_t)262144 * 2);
    _Float16* cc116    = (_Float16*)alloc((size_t)262144 * 2);
    _Float16* w2t      = (_Float16*)alloc((size_t)4096 * 2);
    _Float16* w2c      = (_Float16*)alloc((size_t)4096 * 2);
    float*    b2t      = (float*)alloc(16 * 4);
    float*    b2c      = (float*)alloc(16 * 4);

    _Float16* X     = (_Float16*)alloc((size_t)50331648 * 2);
    _Float16* X1    = (_Float16*)alloc((size_t)50331648 * 2);
    float*    gates = (float*)alloc((size_t)12582912 * 4);
    float*    cbuf  = (float*)alloc((size_t)3145728 * 4);
    _Float16* h1a   = (_Float16*)alloc((size_t)3145728 * 2);
    _Float16* h1b   = (_Float16*)alloc((size_t)3145728 * 2);
    _Float16* Hcat  = (_Float16*)alloc((size_t)6291456 * 2);

    float*    Ef[4]; _Float16* Eh[4];
    const int NK[4]  = {384, 6144, 384, 6144};
    const int SPB[4] = {Tsz, Tsz * Csz, Tsz, Tsz * Csz};
    const int* IDS[4] = {(const int*)d_in[22], (const int*)d_in[24],
                         (const int*)d_in[23], (const int*)d_in[25]};
    for (int k = 0; k < 4; ++k) {
        Ef[k] = (float*)alloc((size_t)NK[k] * Dsz * 4);
        Eh[k] = (_Float16*)alloc((size_t)NK[k] * Dsz * 2);
    }

    _Float16* catbuf = (_Float16*)alloc((size_t)12582912 * 2);
    float*    engbuf = (float*)alloc((size_t)6291456 * 4);
    float*    tblF   = (float*)alloc((size_t)393216 * 4);
    _Float16* tblH   = (_Float16*)alloc((size_t)393216 * 2);
    float*    colF   = (float*)alloc((size_t)6291456 * 4);
    _Float16* colH   = (_Float16*)alloc((size_t)6291456 * 2);
    float*    qF     = (float*)alloc((size_t)393216 * 4);
    float*    kF     = (float*)alloc((size_t)6291456 * 4);
    float*    vF     = (float*)alloc((size_t)6291456 * 4);
    _Float16* ctxH   = (_Float16*)alloc((size_t)393216 * 2);
    float*    projF  = (float*)alloc((size_t)393216 * 4);
    float*    tbl2F  = (float*)alloc((size_t)393216 * 4);
    _Float16* tbl2H  = (_Float16*)alloc((size_t)393216 * 2);
    _Float16* hidT   = (_Float16*)alloc((size_t)98304 * 2);
    _Float16* hidC   = (_Float16*)alloc((size_t)1572864 * 2);
    float*    logT   = (float*)alloc((size_t)6144 * 4);
    float*    logC   = (float*)alloc((size_t)98304 * 4);

    auto cvt = [&](const float* s, _Float16* d, long long n) {
        cvt_f16_kernel<<<(unsigned)((n + 255) / 256), 256, 0, stream>>>(s, d, n);
    };
    cvt(lstm_Wih,   wih16,    16777216LL);
    cvt(lstm_Whh,   whh16,    8388608LL);
    cvt(after_W,    after16,  4194304LL);
    cvt(gate_W,     gate16,   4194304LL);
    cvt(energy_W,   energy16, 4194304LL);
    cvt(attn_in_W,  ain16,    3145728LL);
    cvt(attn_out_W, aout16,   1048576LL);
    cvt(cls_t_W1,   ct116,    262144LL);
    cvt(cls_c_W1,   cc116,    262144LL);
    pad_cls2_kernel<<<16, 256, 0, stream>>>(cls_t_W2, cls_t_b2, w2t, b2t);
    pad_cls2_kernel<<<16, 256, 0, stream>>>(cls_c_W2, cls_c_b2, w2c, b2c);

    auto gemm = [&](const _Float16* A, int lda, const _Float16* W, int ldw,
                    const float* bias, float* Cp, int ldc, int accum,
                    _Float16* Ch, int ldch, int M, int N, int K, int act) {
        dim3 g(N / 64, (M + 127) / 128);
        gemm_wmma_kernel<<<g, 256, 0, stream>>>(A, lda, W, ldw, bias, Cp, ldc,
                                                accum, Ch, ldch, M, N, K, act);
    };

    // ---- span embeddings: gather -> 2-layer BiLSTM -> linear+leaky ----
    for (int k = 0; k < 4; ++k) {
        int Nn = NK[k];
        gather_kernel<<<Nn * Lsz, 256, 0, stream>>>(seq, IDS[k], X, SPB[k]);
        for (int layer = 0; layer < 2; ++layer) {
            const _Float16* Xin = layer ? X1 : X;
            for (int dir = 0; dir < 2; ++dir) {
                size_t wsel = (size_t)((k * 2 + layer) * 2 + dir);
                const _Float16* wih = wih16 + wsel * (size_t)G4 * Dsz;
                const _Float16* whh = whh16 + wsel * (size_t)G4 * Hsz;
                const float*    bb  = lstm_b + wsel * G4;
                _Float16* hptr; int ldh;
                if (layer == 0) { hptr = dir ? h1b : h1a; ldh = Hsz; }
                else            { hptr = Hcat + dir * Hsz; ldh = Dsz; }
                for (int t = 0; t < Lsz; ++t) {
                    int tt = dir ? (Lsz - 1 - t) : t;
                    gemm(Xin + (size_t)tt * Dsz, Lsz * Dsz, wih, Dsz, bb,
                         gates, G4, 0, nullptr, 0, Nn, G4, Dsz, 0);
                    if (t > 0)
                        gemm(hptr, ldh, whh, Hsz, nullptr,
                             gates, G4, 1, nullptr, 0, Nn, G4, Hsz, 0);
                    _Float16* hseq = nullptr; int ldseq = 0;
                    if (layer == 0) {
                        hseq = X1 + (size_t)tt * Dsz + dir * Hsz;
                        ldseq = Lsz * Dsz;
                    }
                    lstm_cell_kernel<<<(Nn * Hsz + 255) / 256, 256, 0, stream>>>(
                        gates, cbuf, hptr, ldh, hseq, ldseq, Nn, t == 0);
                }
            }
        }
        gemm(Hcat, Dsz, after16 + (size_t)k * Dsz * Dsz, Dsz,
             after_b + (size_t)k * Dsz, Ef[k], Dsz, 0, Eh[k], Dsz,
             Nn, Dsz, Dsz, 1);
    }

    // ---- gated pooling ----
    auto gated = [&](int nameI, int commI, int gi, int gj, int ei,
                     float* outF, _Float16* outH) {
        int Nn = NK[nameI];
        gemm(Eh[nameI], Dsz, gate16 + (size_t)gi * Dsz * Dsz, Dsz,
             gate_b + (size_t)gi * Dsz, nullptr, 0, 0, catbuf, 2 * Dsz,
             Nn, Dsz, Dsz, 1);
        gemm(Eh[commI], Dsz, gate16 + (size_t)gj * Dsz * Dsz, Dsz,
             gate_b + (size_t)gj * Dsz, nullptr, 0, 0, catbuf + Dsz, 2 * Dsz,
             Nn, Dsz, Dsz, 1);
        gemm(catbuf, 2 * Dsz, energy16 + (size_t)ei * Dsz * 2 * Dsz, 2 * Dsz,
             energy_b + (size_t)ei * Dsz, engbuf, Dsz, 0, nullptr, 0,
             Nn, Dsz, 2 * Dsz, 0);
        pool_l2n_kernel<<<Nn, 256, 0, stream>>>(Ef[nameI], Ef[commI], engbuf,
                                                outF, outH, 0);
    };
    gated(0, 2, 0, 1, 0, tblF, tblH);
    gated(1, 3, 2, 3, 1, colF, colH);

    // ---- per-table cross attention ----
    gemm(tblH, Dsz, ain16,                       Dsz, attn_in_b,         qF, Dsz, 0, nullptr, 0, 384,  Dsz, Dsz, 0);
    gemm(colH, Dsz, ain16 + (size_t)Dsz * Dsz,   Dsz, attn_in_b + Dsz,   kF, Dsz, 0, nullptr, 0, 6144, Dsz, Dsz, 0);
    gemm(colH, Dsz, ain16 + (size_t)2*Dsz * Dsz, Dsz, attn_in_b + 2*Dsz, vF, Dsz, 0, nullptr, 0, 6144, Dsz, Dsz, 0);
    attn_kernel<<<384 * 8, 128, 0, stream>>>(qF, kF, vF, ctxH);
    gemm(ctxH, Dsz, aout16, Dsz, attn_out_b, projF, Dsz, 0, nullptr, 0,
         384, Dsz, Dsz, 0);
    pool_l2n_kernel<<<384, 256, 0, stream>>>(tblF, projF, nullptr,
                                             tbl2F, tbl2H, 1);

    // ---- classifiers ----
    gemm(tbl2H, Dsz, ct116, Dsz, cls_t_b1, nullptr, 0, 0, hidT, 256,
         384, 256, Dsz, 1);
    gemm(colH, Dsz, cc116, Dsz, cls_c_b1, nullptr, 0, 0, hidC, 256,
         6144, 256, Dsz, 1);
    {
        dim3 gt(1, (384 + 127) / 128);
        gemm_small_kernel<<<gt, 256, 0, stream>>>(hidT, 256, w2t, 256, b2t,
                                                  logT, 16, 384, 256);
        dim3 gc(1, (6144 + 127) / 128);
        gemm_small_kernel<<<gc, 256, 0, stream>>>(hidC, 256, w2c, 256, b2c,
                                                  logC, 16, 6144, 256);
    }

    pack_kernel<<<(Bsz * Tsz * 17 * 2 + 255) / 256, 256, 0, stream>>>(
        logT, logC, (float*)d_out);
}
